// RNN_73186242724432
// MI455X (gfx1250) — matrix operand order repaired and checked by reading, not compiled, not run on hardware
//
#include <hip/hip_runtime.h>

// Shapes from the reference
#define B_  256
#define T_  128
#define I_  512
#define H_  1024
#define O_  128
#define TH_ (T_ * H_)   // 131072

typedef __attribute__((ext_vector_type(16))) __bf16 v16bf;
typedef __attribute__((ext_vector_type(8)))  __bf16 v8bf;
typedef __attribute__((ext_vector_type(8)))  float  v8f;

// ---------------------------------------------------------------------------
// WMMA fragment loaders (v_wmma_f32_16x16x32_bf16, wave32)
//
// A (16x32, M x K), row-major activations A[m][k]:
//   lane L (0-15):  row M=L,  VGPR0-3 hold K=koff..koff+7, VGPR4-7 K=koff+16..+23
//   where koff = 0 for lanes 0-15, 8 for lanes 16-31.
// B (32x16, K x N), weights stored row-major as W[n][k] (n = output col):
//   lane L: column N = L&15, same K order as A (koff..koff+7, koff+16..+23).
// C/D (16x16 f32): VGPR r, lane L -> (M = r + (L>>4)*8, N = L&15).
// ---------------------------------------------------------------------------
__device__ __forceinline__ v16bf load_a_frag(const __bf16* __restrict__ p0,
                                             int ld, int lane) {
  const int r    = lane & 15;
  const int koff = (lane >> 4) << 3;          // 0 or 8
  const __bf16* p = p0 + (size_t)r * ld + koff;
  v8bf lo = *(const v8bf*)(p);
  v8bf hi = *(const v8bf*)(p + 16);
  return __builtin_shufflevector(lo, hi, 0, 1, 2, 3, 4, 5, 6, 7,
                                         8, 9, 10, 11, 12, 13, 14, 15);
}

__device__ __forceinline__ v16bf load_b_frag(const __bf16* __restrict__ p0,
                                             int ld, int lane) {
  const int c    = lane & 15;
  const int koff = (lane >> 4) << 3;          // 0 or 8
  const __bf16* p = p0 + (size_t)c * ld + koff;
  v8bf lo = *(const v8bf*)(p);
  v8bf hi = *(const v8bf*)(p + 16);
  return __builtin_shufflevector(lo, hi, 0, 1, 2, 3, 4, 5, 6, 7,
                                         8, 9, 10, 11, 12, 13, 14, 15);
}

// One wave computes a 32x32 f32 tile: 2x2 grid of 16x16 WMMA accumulators,
// marching K in steps of 32.
__device__ __forceinline__ void wave_gemm_32x32(const __bf16* __restrict__ A, int lda,
                                                const __bf16* __restrict__ Bw, int ldb,
                                                int K, int lane, v8f acc[2][2]) {
  for (int k = 0; k < K; k += 32) {
    v16bf a0 = load_a_frag(A + k, lda, lane);
    v16bf a1 = load_a_frag(A + (size_t)16 * lda + k, lda, lane);
    v16bf b0 = load_b_frag(Bw + k, ldb, lane);
    v16bf b1 = load_b_frag(Bw + (size_t)16 * ldb + k, ldb, lane);
    acc[0][0] = __builtin_amdgcn_wmma_f32_16x16x32_bf16(false, a0, false, b0,
                                                        (short)0, acc[0][0], false, false);
    acc[0][1] = __builtin_amdgcn_wmma_f32_16x16x32_bf16(false, a0, false, b1,
                                                        (short)0, acc[0][1], false, false);
    acc[1][0] = __builtin_amdgcn_wmma_f32_16x16x32_bf16(false, a1, false, b0,
                                                        (short)0, acc[1][0], false, false);
    acc[1][1] = __builtin_amdgcn_wmma_f32_16x16x32_bf16(false, a1, false, b1,
                                                        (short)0, acc[1][1], false, false);
  }
}

// ---------------------------------------------------------------------------
// Conversion / init kernels
// ---------------------------------------------------------------------------
__global__ void k_cvt(const float* __restrict__ s, __bf16* __restrict__ d, int n) {
  int i = blockIdx.x * 256 + threadIdx.x;
  if (i < n) d[i] = (__bf16)s[i];
}

__global__ void k_init_h(const float* __restrict__ hidden, __bf16* __restrict__ h16) {
  int i = blockIdx.x * 256 + threadIdx.x;   // exactly B_*H_ threads
  h16[i] = (__bf16)hidden[i];
}

__global__ void k_init_out(const float* __restrict__ fc_b, float* __restrict__ out,
                           int* __restrict__ bar) {
  int i = blockIdx.x * 256 + threadIdx.x;   // exactly B_*O_ threads
  out[i] = fc_b[i & (O_ - 1)];
  if (i == 0) *bar = 0;
}

// ---------------------------------------------------------------------------
// xw = x @ W_ih^T + b_ih     (M = B*T, N = H, K = I), f32 output
// Block tile 64x64, 128 threads (4 waves of 32x32).
// ---------------------------------------------------------------------------
__global__ void k_gemm_xw(const __bf16* __restrict__ xb, const __bf16* __restrict__ wih,
                          const float* __restrict__ b_ih, float* __restrict__ xw) {
  const int w = threadIdx.x >> 5, lane = threadIdx.x & 31;
  const int m0 = blockIdx.x * 64 + (w & 1) * 32;
  const int n0 = blockIdx.y * 64 + (w >> 1) * 32;
  v8f acc[2][2] = {};
  wave_gemm_32x32(xb + (size_t)m0 * I_, I_, wih + (size_t)n0 * I_, I_, I_, lane, acc);
#pragma unroll
  for (int i = 0; i < 2; ++i)
#pragma unroll
    for (int j = 0; j < 2; ++j) {
      const int n  = n0 + j * 16 + (lane & 15);
      const int mb = m0 + i * 16 + ((lane >> 4) << 3);
      const float bias = b_ih[n];
#pragma unroll
      for (int r = 0; r < 8; ++r)
        xw[(size_t)(mb + r) * H_ + n] = acc[i][j][r] + bias;
    }
}

// ---------------------------------------------------------------------------
// Persistent recurrent kernel: 128 dependent steps, device-wide barrier
// between steps.  Grid = (B/64) x (H/64) = 4 x 16 = 64 blocks, 128 thr.
// W_hh (2MB bf16) + h (0.5MB) stay resident in the 192MB L2.
// ---------------------------------------------------------------------------
__global__ void k_rnn(const float* __restrict__ xw, const __bf16* __restrict__ whh,
                      const float* __restrict__ b_hh, __bf16* __restrict__ h16,
                      __bf16* __restrict__ rout, float* __restrict__ hT,
                      int* __restrict__ bar) {
  const int NBLK = (B_ / 64) * (H_ / 64);   // 64
  const int w = threadIdx.x >> 5, lane = threadIdx.x & 31;
  const int m0 = blockIdx.x * 64 + (w & 1) * 32;
  const int n0 = blockIdx.y * 64 + (w >> 1) * 32;
  const size_t BH = (size_t)B_ * H_;

  for (int t = 0; t < T_; ++t) {
    const __bf16* hin  = h16 + (size_t)(t & 1) * BH;
    __bf16*       hout = h16 + (size_t)((t + 1) & 1) * BH;

    v8f acc[2][2] = {};
    wave_gemm_32x32(hin + (size_t)m0 * H_, H_, whh + (size_t)n0 * H_, H_, H_, lane, acc);

#pragma unroll
    for (int i = 0; i < 2; ++i)
#pragma unroll
      for (int j = 0; j < 2; ++j) {
        const int n  = n0 + j * 16 + (lane & 15);
        const int mb = m0 + i * 16 + ((lane >> 4) << 3);
        const float bias = b_hh[n];
#pragma unroll
        for (int r = 0; r < 8; ++r) {
          const int m = mb + r;
          float v  = acc[i][j][r] + xw[(size_t)m * TH_ + (size_t)t * H_ + n] + bias;
          float hv = tanhf(v);
          __bf16 hb = (__bf16)hv;
          hout[(size_t)m * H_ + n] = hb;
          rout[(size_t)m * TH_ + (size_t)t * H_ + n] = hb;
          if (t == T_ - 1) hT[(size_t)m * H_ + n] = hv;
        }
      }

    // device-wide barrier between timesteps (monotonic counter, no reset)
    __threadfence();
    __syncthreads();
    if (threadIdx.x == 0) {
      __hip_atomic_fetch_add(bar, 1, __ATOMIC_ACQ_REL, __HIP_MEMORY_SCOPE_AGENT);
      const int goal = NBLK * (t + 1);
      while (__hip_atomic_load(bar, __ATOMIC_ACQUIRE, __HIP_MEMORY_SCOPE_AGENT) < goal)
        __builtin_amdgcn_s_sleep(2);
    }
    __syncthreads();
    __threadfence();
  }
}

// ---------------------------------------------------------------------------
// out = r_out.flat @ fc_w^T + fc_b   (M=256, N=128, K=131072)
// Tiny M*N, huge K -> split-K 64 ways, f32 atomic accumulate into d_out
// (bias pre-seeded by k_init_out).
// ---------------------------------------------------------------------------
__global__ void k_fc(const __bf16* __restrict__ rout, const __bf16* __restrict__ fcw,
                     float* __restrict__ out) {
  const int KC = TH_ / 64;                  // 2048 per split
  const int w = threadIdx.x >> 5, lane = threadIdx.x & 31;
  const int m0 = blockIdx.x * 64 + (w & 1) * 32;
  const int n0 = blockIdx.y * 64 + (w >> 1) * 32;
  const size_t k0 = (size_t)blockIdx.z * KC;
  v8f acc[2][2] = {};
  wave_gemm_32x32(rout + (size_t)m0 * TH_ + k0, TH_,
                  fcw  + (size_t)n0 * TH_ + k0, TH_, KC, lane, acc);
#pragma unroll
  for (int i = 0; i < 2; ++i)
#pragma unroll
    for (int j = 0; j < 2; ++j) {
      const int n  = n0 + j * 16 + (lane & 15);
      const int mb = m0 + i * 16 + ((lane >> 4) << 3);
#pragma unroll
      for (int r = 0; r < 8; ++r)
        atomicAdd(out + (size_t)(mb + r) * O_ + n, acc[i][j][r]);
    }
}

// ---------------------------------------------------------------------------
extern "C" void kernel_launch(void* const* d_in, const int* in_sizes, int n_in,
                              void* d_out, int out_size, void* d_ws, size_t ws_size,
                              hipStream_t stream) {
  const float* x      = (const float*)d_in[0];
  const float* hidden = (const float*)d_in[1];
  const float* W_ih   = (const float*)d_in[2];
  const float* W_hh   = (const float*)d_in[3];
  const float* b_ih   = (const float*)d_in[4];
  const float* b_hh   = (const float*)d_in[5];
  const float* fc_w   = (const float*)d_in[6];
  const float* fc_b   = (const float*)d_in[7];

  float* out = (float*)d_out;          // [B,O] then [1,B,H] concatenated
  float* hT  = out + B_ * O_;

  char* base = (char*)d_ws;
  size_t off = 0;
  auto take = [&](size_t bytes) -> void* {
    void* p = base + off;
    off += (bytes + 255) & ~(size_t)255;
    return p;
  };
  __bf16* xb   = (__bf16*)take((size_t)B_ * T_ * I_ * 2);
  __bf16* wih  = (__bf16*)take((size_t)H_ * I_ * 2);
  __bf16* whh  = (__bf16*)take((size_t)H_ * H_ * 2);
  __bf16* fcw  = (__bf16*)take((size_t)O_ * TH_ * 2);
  float*  xw   = (float*) take((size_t)B_ * TH_ * 4);
  __bf16* rout = (__bf16*)take((size_t)B_ * TH_ * 2);
  __bf16* h16  = (__bf16*)take((size_t)2 * B_ * H_ * 2);
  int*    bar  = (int*)   take(256);

  // bf16 conversions of inputs / weights (layouts stay row-major: W[n][k]
  // is exactly the WMMA B-fragment layout, no transpose needed)
  k_cvt<<<(B_ * T_ * I_) / 256, 256, 0, stream>>>(x, xb, B_ * T_ * I_);
  k_cvt<<<(H_ * I_) / 256, 256, 0, stream>>>(W_ih, wih, H_ * I_);
  k_cvt<<<(H_ * H_) / 256, 256, 0, stream>>>(W_hh, whh, H_ * H_);
  k_cvt<<<(O_ * TH_) / 256, 256, 0, stream>>>(fc_w, fcw, O_ * TH_);
  k_init_h<<<(B_ * H_) / 256, 256, 0, stream>>>(hidden, h16);
  k_init_out<<<(B_ * O_) / 256, 256, 0, stream>>>(fc_b, out, bar);

  // 1) xw = x @ W_ih^T + b_ih
  k_gemm_xw<<<dim3((B_ * T_) / 64, H_ / 64), 128, 0, stream>>>(xb, wih, b_ih, xw);
  // 2) persistent 128-step recurrence with device-wide barrier
  k_rnn<<<dim3(B_ / 64, H_ / 64), 128, 0, stream>>>(xw, whh, b_hh, h16, rout, hT, bar);
  // 3) split-K FC into pre-biased d_out
  k_fc<<<dim3(B_ / 64, O_ / 64, 64), 128, 0, stream>>>(rout, fcw, out);
}
